// DenseFlashBlock_20014547599381
// MI455X (gfx1250) — matrix operand order repaired and checked by reading, not compiled, not run on hardware
//
#include <hip/hip_runtime.h>
#include <hip/hip_bf16.h>

typedef __attribute__((ext_vector_type(16))) __bf16 v16bf;
typedef __attribute__((ext_vector_type(8)))  float  v8f;
typedef int v4i32 __attribute__((vector_size(16)));

constexpr int BATCH  = 4;
constexpr int SEQ    = 1024;
constexpr int DMODEL = 1280;
constexpr int NH     = 20;
constexpr int DHEAD  = 64;
constexpr int TD     = 3 * DMODEL;   // qkv row stride
constexpr int ROWS   = BATCH * SEQ;  // 4096

#define DEV_INLINE __device__ __forceinline__

#if defined(__has_builtin)
# if __has_builtin(__builtin_amdgcn_global_load_async_to_lds_b128)
#  define HAVE_ASYNC_BUILTIN 1
# endif
# if __has_builtin(__builtin_amdgcn_s_wait_asynccnt)
#  define HAVE_WAIT_ASYNC_BUILTIN 1
# endif
#endif

DEV_INLINE v8f wmma_bf16(v16bf a, v16bf b, v8f c) {
  return __builtin_amdgcn_wmma_f32_16x16x32_bf16(false, a, false, b, (short)0, c, false, false);
}

// Async 16-byte global -> LDS copy (per lane), tracked by ASYNCcnt.
DEV_INLINE void cp_async16(__bf16* lds, const __bf16* g) {
#if defined(HAVE_ASYNC_BUILTIN)
  __builtin_amdgcn_global_load_async_to_lds_b128(
      (__attribute__((address_space(1))) v4i32*)(uintptr_t)g,
      (__attribute__((address_space(3))) v4i32*)(uint32_t)(uintptr_t)lds, 0, 0);
#else
  // flat shared address: low 32 bits are the LDS byte offset (aperture in high bits)
  uint32_t laddr = (uint32_t)(uintptr_t)lds;
  uint64_t gaddr = (uint64_t)(uintptr_t)g;
  asm volatile("global_load_async_to_lds_b128 %0, %1, off"
               :: "v"(laddr), "v"(gaddr) : "memory");
#endif
}

DEV_INLINE void wait_async0() {
#if defined(HAVE_WAIT_ASYNC_BUILTIN)
  __builtin_amdgcn_s_wait_asynccnt(0);
#else
  asm volatile("s_wait_asynccnt 0x0" ::: "memory");
#endif
}

DEV_INLINE float gelu_tanh(float x) {
  const float k0 = 0.7978845608028654f;   // sqrt(2/pi)
  const float k1 = 0.044715f;
  float t = tanhf(k0 * (x + k1 * x * x * x));
  return 0.5f * x * (1.0f + t);
}

// ---------------------------------------------------------------------------
// Weight convert: fp32 (K x N) row-major  ->  bf16 (N x K) row-major
// ---------------------------------------------------------------------------
__global__ __launch_bounds__(256)
void k_transpose_w(const float* __restrict__ in, __bf16* __restrict__ out, int K, int N) {
  __shared__ float tile[32][33];
  const int tx = threadIdx.x & 31, ty = threadIdx.x >> 5;
  const int nb = blockIdx.x * 32, kb = blockIdx.y * 32;
  #pragma unroll
  for (int i = ty; i < 32; i += 8)
    tile[i][tx] = in[(size_t)(kb + i) * N + nb + tx];
  __syncthreads();
  #pragma unroll
  for (int i = ty; i < 32; i += 8)
    out[(size_t)(nb + i) * K + kb + tx] = (__bf16)tile[tx][i];
}

// ---------------------------------------------------------------------------
// LayerNorm fp32 row -> bf16 row (one block per row)
// ---------------------------------------------------------------------------
__global__ __launch_bounds__(256)
void k_ln_bf16(const float* __restrict__ x, const float* __restrict__ g,
               const float* __restrict__ bb, __bf16* __restrict__ out) {
  const int row = blockIdx.x, tid = threadIdx.x;
  const float* p = x + (size_t)row * DMODEL;
  float s = 0.f, ss = 0.f;
  for (int i = tid; i < DMODEL; i += 256) { float v = p[i]; s += v; ss += v * v; }
  __shared__ float r1[256], r2[256];
  r1[tid] = s; r2[tid] = ss; __syncthreads();
  for (int o = 128; o > 0; o >>= 1) {
    if (tid < o) { r1[tid] += r1[tid + o]; r2[tid] += r2[tid + o]; }
    __syncthreads();
  }
  const float mu  = r1[0] * (1.0f / DMODEL);
  const float var = r2[0] * (1.0f / DMODEL) - mu * mu;
  const float inv = rsqrtf(var + 1e-5f);
  for (int i = tid; i < DMODEL; i += 256)
    out[(size_t)row * DMODEL + i] = (__bf16)((p[i] - mu) * inv * g[i] + bb[i]);
}

// h = a + y (store fp32), z = LN(h) -> bf16
__global__ __launch_bounds__(256)
void k_add_ln(const float* __restrict__ a, const float* __restrict__ y,
              const float* __restrict__ g, const float* __restrict__ bb,
              float* __restrict__ h, __bf16* __restrict__ z) {
  const int row = blockIdx.x, tid = threadIdx.x;
  const float* pa = a + (size_t)row * DMODEL;
  const float* py = y + (size_t)row * DMODEL;
  float* ph = h + (size_t)row * DMODEL;
  float s = 0.f, ss = 0.f;
  for (int i = tid; i < DMODEL; i += 256) {
    float v = pa[i] + py[i]; ph[i] = v; s += v; ss += v * v;
  }
  __shared__ float r1[256], r2[256];
  r1[tid] = s; r2[tid] = ss; __syncthreads();
  for (int o = 128; o > 0; o >>= 1) {
    if (tid < o) { r1[tid] += r1[tid + o]; r2[tid] += r2[tid + o]; }
    __syncthreads();
  }
  const float mu  = r1[0] * (1.0f / DMODEL);
  const float var = r2[0] * (1.0f / DMODEL) - mu * mu;
  const float inv = rsqrtf(var + 1e-5f);
  for (int i = tid; i < DMODEL; i += 256)
    z[(size_t)row * DMODEL + i] = (__bf16)((ph[i] - mu) * inv * g[i] + bb[i]);
}

// ---------------------------------------------------------------------------
// Generic WMMA GEMM:  C(MxN) = A(MxK, bf16 row-major) * Bt(NxK, bf16)^T + bias
// Block tile 128x128, K-step 32, 8 waves (4x2), wave tile 32x64.
// Async global->LDS staging, double-buffered. LDS rows padded to 40 elements
// (80B = 20 banks) for conflict-free fragment reads.
// ---------------------------------------------------------------------------
template<bool OUT_BF16, bool GELU, bool RESID>
__global__ __launch_bounds__(256)
void k_gemm_wmma(const __bf16* __restrict__ A, const __bf16* __restrict__ Bt,
                 const float* __restrict__ bias, const float* __restrict__ resid,
                 void* __restrict__ Cout, int M, int N, int K) {
  __shared__ __bf16 sA[2][128][40];
  __shared__ __bf16 sB[2][128][40];
  const int tid  = threadIdx.x;
  const int lane = tid & 31;
  const int wave = tid >> 5;
  const int wm   = wave >> 1;          // 0..3  -> M offset 32*wm
  const int wn   = wave & 1;           // 0..1  -> N offset 64*wn
  const int half = lane >> 4;          // 0/1
  const int l15  = lane & 15;
  const int m0 = blockIdx.x * 128;
  const int n0 = blockIdx.y * 128;

  // staging coordinates for this thread: 2 chunks of 8 bf16 per matrix
  const int c0r = tid >> 2,         c0c = (tid & 3) * 8;
  const int c1r = (tid + 256) >> 2, c1c = c0c;

  v8f acc[2][4] = {};

  auto stage = [&](int buf, int k0) {
    cp_async16(&sA[buf][c0r][c0c], &A [(size_t)(m0 + c0r) * K + k0 + c0c]);
    cp_async16(&sB[buf][c0r][c0c], &Bt[(size_t)(n0 + c0r) * K + k0 + c0c]);
    cp_async16(&sA[buf][c1r][c1c], &A [(size_t)(m0 + c1r) * K + k0 + c1c]);
    cp_async16(&sB[buf][c1r][c1c], &Bt[(size_t)(n0 + c1r) * K + k0 + c1c]);
  };

  stage(0, 0);
  int buf = 0;
  for (int k0 = 0; k0 < K; k0 += 32) {
    wait_async0();        // my staged loads for `buf` are done
    __syncthreads();      // => everyone's staged loads for `buf` are done
    if (k0 + 32 < K) stage(buf ^ 1, k0 + 32);   // overlap next tile with compute

    v16bf af[2], bfr[4];
    #pragma unroll
    for (int i = 0; i < 2; ++i) {        // A frag: lane&15 = row, K base (lane>>4)*8, +16
      const __bf16* p = &sA[buf][wm * 32 + i * 16 + l15][half * 8];
      *(uint4*)&af[i]       = *(const uint4*)p;
      *((uint4*)&af[i] + 1) = *(const uint4*)(p + 16);
    }
    #pragma unroll
    for (int j = 0; j < 4; ++j) {        // B frag: lane&15 = col(n), K base (lane>>4)*8, +16
      const __bf16* p = &sB[buf][wn * 64 + j * 16 + l15][half * 8];
      *(uint4*)&bfr[j]       = *(const uint4*)p;
      *((uint4*)&bfr[j] + 1) = *(const uint4*)(p + 16);
    }
    #pragma unroll
    for (int i = 0; i < 2; ++i)
      #pragma unroll
      for (int j = 0; j < 4; ++j)
        acc[i][j] = wmma_bf16(af[i], bfr[j], acc[i][j]);
    buf ^= 1;
  }

  // epilogue: C layout — VGPR r: lanes0-15 M=r, lanes16-31 M=r+8; N = lane&15
  #pragma unroll
  for (int i = 0; i < 2; ++i) {
    const int mb = m0 + wm * 32 + i * 16 + half * 8;
    #pragma unroll
    for (int j = 0; j < 4; ++j) {
      const int col = n0 + wn * 64 + j * 16 + l15;
      const float bv = bias[col];
      #pragma unroll
      for (int r = 0; r < 8; ++r) {
        const size_t idx = (size_t)(mb + r) * N + col;
        float v = acc[i][j][r] + bv;
        if (GELU)  v = gelu_tanh(v);
        if (RESID) v += resid[idx];
        if (OUT_BF16) ((__bf16*)Cout)[idx] = (__bf16)v;
        else          ((float*)Cout)[idx]  = v;
      }
    }
  }
}

// ---------------------------------------------------------------------------
// V transpose: qkv bf16 (b,s,3D) v-part -> vt[b][h][dh][s]
// ---------------------------------------------------------------------------
__global__ __launch_bounds__(256)
void k_v_transpose(const __bf16* __restrict__ qkv, __bf16* __restrict__ vt) {
  __shared__ __bf16 t[32][33];
  const int bh = blockIdx.z, b = bh / NH, h = bh % NH;
  const int s0 = blockIdx.x * 32, d0 = blockIdx.y * 32;
  const int tx = threadIdx.x & 31, ty = threadIdx.x >> 5;
  #pragma unroll
  for (int i = ty; i < 32; i += 8)
    t[i][tx] = qkv[(size_t)(b * SEQ + s0 + i) * TD + 2 * DMODEL + h * DHEAD + d0 + tx];
  __syncthreads();
  #pragma unroll
  for (int i = ty; i < 32; i += 8)
    vt[((size_t)bh * DHEAD + d0 + i) * SEQ + s0 + tx] = t[tx][i];
}

// ---------------------------------------------------------------------------
// Flash attention: one wave per 16 query rows; 4 waves/block; grid (S/64, B*H)
// ---------------------------------------------------------------------------
__global__ __launch_bounds__(128)
void k_attention(const __bf16* __restrict__ qkv, const __bf16* __restrict__ vt,
                 const unsigned char* __restrict__ amask, __bf16* __restrict__ yout) {
  __shared__ __bf16 pbuf[4][16][40];   // per-wave 16x32 P tile (+pad)
  const int lane = threadIdx.x & 31;
  const int wv   = threadIdx.x >> 5;
  const int half = lane >> 4;
  const int l15  = lane & 15;
  const int bh = blockIdx.y, b = bh / NH, h = bh % NH;
  const int q0 = blockIdx.x * 64 + wv * 16;

  const __bf16* qbase = qkv + (size_t)b * SEQ * TD + h * DHEAD;
  const __bf16* kbase = qbase + DMODEL;
  const __bf16* vbase = vt + (size_t)bh * DHEAD * SEQ;

  // Q fragments (A layout): dh blocks [0,32) and [32,64)
  v16bf qf[2];
  {
    const __bf16* qp = qbase + (size_t)(q0 + l15) * TD + half * 8;
    #pragma unroll
    for (int i = 0; i < 2; ++i) {
      *(uint4*)&qf[i]       = *(const uint4*)(qp + i * 32);
      *((uint4*)&qf[i] + 1) = *(const uint4*)(qp + i * 32 + 16);
    }
  }

  v8f oacc[4] = {};
  float mrow[8], lrow[8];
  #pragma unroll
  for (int r = 0; r < 8; ++r) { mrow[r] = -3.0e38f; lrow[r] = 0.0f; }

  const int nk = q0 + 16;              // causal: keys [0, nk)
  for (int kt = 0; kt < nk; kt += 32) {
    const bool t1ok = (kt + 16) < nk;
    v8f sacc[2] = {};
    #pragma unroll
    for (int t = 0; t < 2; ++t) {
      if (t == 0 || t1ok) {
        const __bf16* kp = kbase + (size_t)(kt + t * 16 + l15) * TD + half * 8;
        v8f s = {};
        #pragma unroll
        for (int i = 0; i < 2; ++i) {
          v16bf kf;
          *(uint4*)&kf       = *(const uint4*)(kp + i * 32);
          *((uint4*)&kf + 1) = *(const uint4*)(kp + i * 32 + 16);
          s = wmma_bf16(qf[i], kf, s);
        }
        sacc[t] = s;
      }
    }

    // scale + causal/key mask, online softmax stats (rows live in 16-lane halves)
    float p0[8], p1[8], rmax[8];
    const int qrb = q0 + half * 8;
    const int k0c = kt + l15, k1c = kt + 16 + l15;
    const bool am0 = amask[b * SEQ + k0c] != 0;
    const bool am1 = t1ok ? (amask[b * SEQ + k1c] != 0) : false;
    #pragma unroll
    for (int r = 0; r < 8; ++r) {
      const int qq = qrb + r;
      float s0 = sacc[0][r] * 0.125f;
      float s1 = sacc[1][r] * 0.125f;
      if (!(k0c <= qq) || !am0) s0 = -3.0e38f;
      if (!t1ok || !(k1c <= qq) || !am1) s1 = -3.0e38f;
      p0[r] = s0; p1[r] = s1;
      rmax[r] = fmaxf(s0, s1);
    }
    #pragma unroll
    for (int r = 0; r < 8; ++r)
      #pragma unroll
      for (int m = 1; m < 16; m <<= 1)
        rmax[r] = fmaxf(rmax[r], __shfl_xor(rmax[r], m));

    float csc[8];
    #pragma unroll
    for (int r = 0; r < 8; ++r) {
      const float nm = fmaxf(mrow[r], rmax[r]);
      csc[r] = __expf(mrow[r] - nm);
      mrow[r] = nm;
      p0[r] = __expf(p0[r] - nm);
      p1[r] = __expf(p1[r] - nm);
      float rs = p0[r] + p1[r];
      #pragma unroll
      for (int m = 1; m < 16; m <<= 1) rs += __shfl_xor(rs, m);
      lrow[r] = lrow[r] * csc[r] + rs;
    }
    #pragma unroll
    for (int j = 0; j < 4; ++j)
      #pragma unroll
      for (int r = 0; r < 8; ++r)
        oacc[j][r] *= csc[r];

    // P: C-layout -> A-layout through per-wave LDS
    #pragma unroll
    for (int r = 0; r < 8; ++r) {
      pbuf[wv][half * 8 + r][l15]      = (__bf16)p0[r];
      pbuf[wv][half * 8 + r][16 + l15] = (__bf16)p1[r];
    }
    __threadfence_block();
    v16bf pf;
    {
      const __bf16* pp = &pbuf[wv][l15][half * 8];
      *(uint4*)&pf       = *(const uint4*)pp;
      *((uint4*)&pf + 1) = *(const uint4*)(pp + 16);
    }
    __threadfence_block();

    // O += P x V  (V pre-transposed: rows = dh, contiguous in s)
    #pragma unroll
    for (int j = 0; j < 4; ++j) {
      v16bf vf;
      const __bf16* vp = vbase + (size_t)(j * 16 + l15) * SEQ + kt + half * 8;
      *(uint4*)&vf       = *(const uint4*)vp;
      *((uint4*)&vf + 1) = *(const uint4*)(vp + 16);
      oacc[j] = wmma_bf16(pf, vf, oacc[j]);
    }
  }

  #pragma unroll
  for (int j = 0; j < 4; ++j)
    #pragma unroll
    for (int r = 0; r < 8; ++r) {
      const int qq = q0 + half * 8 + r;
      yout[(size_t)(b * SEQ + qq) * DMODEL + h * DHEAD + j * 16 + l15] =
          (__bf16)(oacc[j][r] / lrow[r]);
    }
}

// ---------------------------------------------------------------------------
extern "C" void kernel_launch(void* const* d_in, const int* in_sizes, int n_in,
                              void* d_out, int out_size, void* d_ws, size_t ws_size,
                              hipStream_t stream) {
  const float* hidden        = (const float*)d_in[0];
  const unsigned char* amask = (const unsigned char*)d_in[1];
  const float* ln1_g  = (const float*)d_in[2];
  const float* ln1_b  = (const float*)d_in[3];
  const float* w_qkv  = (const float*)d_in[4];
  const float* b_qkv  = (const float*)d_in[5];
  const float* w_proj = (const float*)d_in[6];
  const float* b_proj = (const float*)d_in[7];
  const float* ln2_g  = (const float*)d_in[8];
  const float* ln2_b  = (const float*)d_in[9];
  const float* w_fc   = (const float*)d_in[10];
  const float* b_fc   = (const float*)d_in[11];
  const float* w_fc2  = (const float*)d_in[12];
  const float* b_fc2  = (const float*)d_in[13];
  float* out = (float*)d_out;

  // workspace carve (256B aligned)
  size_t off = 0;
  auto carve = [&](size_t bytes) -> char* {
    char* p = (char*)d_ws + off;
    off = (off + bytes + 255) & ~(size_t)255;
    return p;
  };
  __bf16* xln    = (__bf16*)carve((size_t)ROWS * DMODEL * 2);
  __bf16* qkvb   = (__bf16*)carve((size_t)ROWS * TD * 2);
  __bf16* vtb    = (__bf16*)carve((size_t)BATCH * NH * DHEAD * SEQ * 2);
  __bf16* ybf    = (__bf16*)carve((size_t)ROWS * DMODEL * 2);
  float*  yproj  = (float*) carve((size_t)ROWS * DMODEL * 4);
  float*  hbuf   = (float*) carve((size_t)ROWS * DMODEL * 4);
  __bf16* zbf    = (__bf16*)carve((size_t)ROWS * DMODEL * 2);
  __bf16* fc1b   = (__bf16*)carve((size_t)ROWS * 4 * DMODEL * 2);
  __bf16* wqkvT  = (__bf16*)carve((size_t)3 * DMODEL * DMODEL * 2);
  __bf16* wprojT = (__bf16*)carve((size_t)DMODEL * DMODEL * 2);
  __bf16* wfcT   = (__bf16*)carve((size_t)4 * DMODEL * DMODEL * 2);
  __bf16* wfc2T  = (__bf16*)carve((size_t)DMODEL * 4 * DMODEL * 2);

  // 1. weights -> bf16 transposed (NxK)
  k_transpose_w<<<dim3(3 * DMODEL / 32, DMODEL / 32), 256, 0, stream>>>(w_qkv,  wqkvT,  DMODEL, 3 * DMODEL);
  k_transpose_w<<<dim3(DMODEL / 32,     DMODEL / 32), 256, 0, stream>>>(w_proj, wprojT, DMODEL, DMODEL);
  k_transpose_w<<<dim3(4 * DMODEL / 32, DMODEL / 32), 256, 0, stream>>>(w_fc,   wfcT,   DMODEL, 4 * DMODEL);
  k_transpose_w<<<dim3(DMODEL / 32, 4 * DMODEL / 32), 256, 0, stream>>>(w_fc2,  wfc2T,  4 * DMODEL, DMODEL);

  // 2. LN1 -> bf16
  k_ln_bf16<<<ROWS, 256, 0, stream>>>(hidden, ln1_g, ln1_b, xln);

  // 3. QKV GEMM (4096 x 3840 x 1280), bf16 out
  k_gemm_wmma<true, false, false><<<dim3(ROWS / 128, 3 * DMODEL / 128), 256, 0, stream>>>(
      xln, wqkvT, b_qkv, nullptr, qkvb, ROWS, 3 * DMODEL, DMODEL);

  // 4. V transpose for PxV B-fragments
  k_v_transpose<<<dim3(SEQ / 32, DHEAD / 32, BATCH * NH), 256, 0, stream>>>(qkvb, vtb);

  // 5. causal flash attention
  k_attention<<<dim3(SEQ / 64, BATCH * NH), 128, 0, stream>>>(qkvb, vtb, amask, ybf);

  // 6. output projection (fp32 out)
  k_gemm_wmma<false, false, false><<<dim3(ROWS / 128, DMODEL / 128), 256, 0, stream>>>(
      ybf, wprojT, b_proj, nullptr, yproj, ROWS, DMODEL, DMODEL);

  // 7. h = hidden + yproj ; z = LN2(h) -> bf16
  k_add_ln<<<ROWS, 256, 0, stream>>>(hidden, yproj, ln2_g, ln2_b, hbuf, zbf);

  // 8. FC1 + GELU (bf16 out)
  k_gemm_wmma<true, true, false><<<dim3(ROWS / 128, 4 * DMODEL / 128), 256, 0, stream>>>(
      zbf, wfcT, b_fc, nullptr, fc1b, ROWS, 4 * DMODEL, DMODEL);

  // 9. FC2 + residual h -> final fp32 output
  k_gemm_wmma<false, false, true><<<dim3(ROWS / 128, DMODEL / 128), 256, 0, stream>>>(
      fc1b, wfc2T, b_fc2, hbuf, out, ROWS, DMODEL, 4 * DMODEL);

  (void)in_sizes; (void)n_in; (void)out_size; (void)ws_size;
}
